// Attention_11175504904574
// MI455X (gfx1250) — matrix operand order repaired and checked by reading, not compiled, not run on hardware
//
#include <hip/hip_runtime.h>
#include <hip/hip_bf16.h>
#include <math.h>

// ---------------------------------------------------------------------------
// Attention layer for MI455X (gfx1250): bf16 WMMA everywhere, fp32 accum.
// BS=2, SEQ=2048, DIM=2048, H=16, KVH=8, HD=128.
// ---------------------------------------------------------------------------

#define BSZ   2
#define SEQL  2048
#define DIM   2048
#define NH    16
#define NKV   8
#define HD    128

typedef __bf16 bf16x8 __attribute__((ext_vector_type(8)));
typedef __bf16 v16bf  __attribute__((ext_vector_type(16)));
typedef float  v8f    __attribute__((ext_vector_type(8)));

// ---- WMMA fragment loaders (NT GEMM, both operands row-major over K) ------
// A-matrix 16x32 bf16 layout: lane r=lane&15 -> row; half=lane>>4;
//   VGPR0-3 = K[half*8 .. +7], VGPR4-7 = K[16+half*8 .. +7]  (two b128 loads)
__device__ __forceinline__ v16bf load_frag_a(const __bf16* base, int row0,
                                             int lda, int k0, int lane) {
  const int r = lane & 15, h = lane >> 4;
  const __bf16* p = base + (size_t)(row0 + r) * lda + k0 + h * 8;
  union { v16bf v; bf16x8 c[2]; } u;
  u.c[0] = *(const bf16x8*)(p);
  u.c[1] = *(const bf16x8*)(p + 16);
  return u.v;
}
// B-matrix 32x16 bf16 layout: lane column n=lane&15 -> row of (N,K) operand;
//   VGPR0-3 = K[half*8 .. +7], VGPR4-7 = K[16+half*8 .. +7]  (two b128 loads)
__device__ __forceinline__ v16bf load_frag_b(const __bf16* base, int n0,
                                             int ldb, int k0, int lane) {
  const int r = lane & 15, h = lane >> 4;
  const __bf16* p = base + (size_t)(n0 + r) * ldb + k0 + h * 8;
  union { v16bf v; bf16x8 c[2]; } u;
  u.c[0] = *(const bf16x8*)(p);
  u.c[1] = *(const bf16x8*)(p + 16);
  return u.v;
}

__device__ __forceinline__ v8f wmma_bf16(v16bf a, v16bf b, v8f c) {
  return __builtin_amdgcn_wmma_f32_16x16x32_bf16(false, a, false, b,
                                                 (short)0, c, false, false);
}

// ---------------------------------------------------------------------------
__global__ void cast_f32_bf16(const float* __restrict__ src,
                              __bf16* __restrict__ dst, size_t n) {
  size_t i = (size_t)blockIdx.x * blockDim.x + threadIdx.x;
  size_t stride = (size_t)gridDim.x * blockDim.x;
  for (; i < n; i += stride) dst[i] = (__bf16)src[i];
}

// ---------------------------------------------------------------------------
// C[M,N] = A[M,K] @ B[N,K]^T ; A,B bf16 row-major; C fp32 or bf16 row-major.
// Block = 256 threads (8 waves, 4x2), WG tile 128Mx128N, wave tile 32Mx64N.
// Register double-buffered: next k-step's 12 b128 loads fly over this step's
// 8 WMMAs (all operands L2-resident: working set ~105MB < 192MB L2).
template <bool OUT_F32>
__global__ __launch_bounds__(256) void gemm_nt(const __bf16* __restrict__ A,
                                               const __bf16* __restrict__ Bm,
                                               void* __restrict__ Cout,
                                               int M, int N, int K) {
  const int lane = threadIdx.x & 31;
  const int wave = threadIdx.x >> 5;
  const int r = lane & 15, half = lane >> 4;
  const int m_base = blockIdx.y * 128 + (wave >> 1) * 32;
  const int n_base = blockIdx.x * 128 + (wave & 1) * 64;

  v8f acc[2][4];
#pragma unroll
  for (int mi = 0; mi < 2; ++mi)
#pragma unroll
    for (int ni = 0; ni < 4; ++ni)
#pragma unroll
      for (int v = 0; v < 8; ++v) acc[mi][ni][v] = 0.0f;

  v16bf af[2][2], bf[2][4];
#pragma unroll
  for (int mi = 0; mi < 2; ++mi)
    af[0][mi] = load_frag_a(A, m_base + mi * 16, K, 0, lane);
#pragma unroll
  for (int ni = 0; ni < 4; ++ni)
    bf[0][ni] = load_frag_b(Bm, n_base + ni * 16, K, 0, lane);

  for (int k0 = 0; k0 < K; k0 += 32) {
    const int cur = (k0 >> 5) & 1, nxt = cur ^ 1;
    if (k0 + 32 < K) {
#pragma unroll
      for (int mi = 0; mi < 2; ++mi)
        af[nxt][mi] = load_frag_a(A, m_base + mi * 16, K, k0 + 32, lane);
#pragma unroll
      for (int ni = 0; ni < 4; ++ni)
        bf[nxt][ni] = load_frag_b(Bm, n_base + ni * 16, K, k0 + 32, lane);
    }
#pragma unroll
    for (int mi = 0; mi < 2; ++mi)
#pragma unroll
      for (int ni = 0; ni < 4; ++ni)
        acc[mi][ni] = wmma_bf16(af[cur][mi], bf[cur][ni], acc[mi][ni]);
  }

  // C layout: lane half selects M+=8; VGPR v -> M=v; lane&15 -> N.
#pragma unroll
  for (int mi = 0; mi < 2; ++mi)
#pragma unroll
    for (int ni = 0; ni < 4; ++ni) {
      const int m = m_base + mi * 16 + half * 8;
      const int n = n_base + ni * 16 + r;
#pragma unroll
      for (int v = 0; v < 8; ++v) {
        const size_t off = (size_t)(m + v) * N + n;
        if (OUT_F32) ((float*)Cout)[off] = acc[mi][ni][v];
        else         ((__bf16*)Cout)[off] = (__bf16)acc[mi][ni][v];
      }
    }
}

// ---------------------------------------------------------------------------
// RoPE + relayout: in [B,S,H,HD] -> out [B,H,S,HD] (bf16), interleaved pairs.
__global__ void rope_relayout(const __bf16* __restrict__ tin,
                              const float* __restrict__ cosp,
                              const float* __restrict__ sinp,
                              __bf16* __restrict__ tout, int H) {
  const int hd2 = HD / 2;
  const size_t n = (size_t)BSZ * SEQL * H * hd2;
  const size_t stride = (size_t)gridDim.x * blockDim.x;
  for (size_t p = (size_t)blockIdx.x * blockDim.x + threadIdx.x; p < n;
       p += stride) {
    const int i = (int)(p % hd2);
    size_t t = p / hd2;
    const int h = (int)(t % H);  t /= H;
    const int s = (int)(t % SEQL);
    const int b = (int)(t / SEQL);
    const size_t in_off = (((size_t)b * SEQL + s) * H + h) * HD + 2 * i;
    const float re = (float)tin[in_off];
    const float im = (float)tin[in_off + 1];
    const float c = cosp[s * hd2 + i];
    const float sn = sinp[s * hd2 + i];
    const size_t out_off = (((size_t)b * H + h) * SEQL + s) * HD + 2 * i;
    tout[out_off]     = (__bf16)(re * c - im * sn);
    tout[out_off + 1] = (__bf16)(re * sn + im * c);
  }
}

// V transpose: [B,S,KVH,HD] -> [B,KVH,HD,S] so P@V runs as an NT GEMM.
__global__ void transpose_v(const __bf16* __restrict__ vin,
                            __bf16* __restrict__ vout) {
  const size_t n = (size_t)BSZ * SEQL * NKV * HD;
  const size_t stride = (size_t)gridDim.x * blockDim.x;
  for (size_t p = (size_t)blockIdx.x * blockDim.x + threadIdx.x; p < n;
       p += stride) {
    const int d = (int)(p % HD);
    size_t t = p / HD;
    const int h = (int)(t % NKV);  t /= NKV;
    const int s = (int)(t % SEQL);
    const int b = (int)(t / SEQL);
    vout[(((size_t)b * NKV + h) * HD + d) * SEQL + s] = vin[p];
  }
}

// ---------------------------------------------------------------------------
// Flash attention: one wave (32 threads) per 16-query tile per head.
// Q [B,H,S,HD], K [B,KVH,S,HD], Vt [B,KVH,HD,S] -> O [B,S,H,HD] (all bf16).
// Per 32-key block: K and V fragment loads are issued together up front; the
// V loads hide under the online-softmax VALU work before the PV WMMAs.
__global__ __launch_bounds__(32) void flash_attn(const __bf16* __restrict__ Qh,
                                                 const __bf16* __restrict__ Kh,
                                                 const __bf16* __restrict__ Vt,
                                                 __bf16* __restrict__ Oout) {
  __shared__ __bf16 pl[16 * 32];  // P tile staging (C-layout -> A-fragment)
  const int lane = threadIdx.x & 31;
  const int r = lane & 15, half = lane >> 4;
  const int bid = blockIdx.x;
  const int qt = bid & (SEQL / 16 - 1);
  const int h  = (bid / (SEQL / 16)) & (NH - 1);
  const int b  = bid / ((SEQL / 16) * NH);
  const int kvh = h >> 1;  // NH/NKV = 2 (GQA repeat)
  const int q_base = qt * 16;

  const __bf16* Qbase = Qh + (((size_t)b * NH + h) * SEQL + q_base) * HD;
  const __bf16* Kbase = Kh + (((size_t)b * NKV + kvh) * SEQL) * HD;
  const __bf16* Vbase = Vt + (((size_t)b * NKV + kvh) * HD) * SEQL;

  v16bf qf[4];
#pragma unroll
  for (int c = 0; c < 4; ++c) qf[c] = load_frag_a(Qbase, 0, HD, c * 32, lane);

  v8f o[8];
  float mrow[8], lrow[8];
#pragma unroll
  for (int d = 0; d < 8; ++d)
#pragma unroll
    for (int v = 0; v < 8; ++v) o[d][v] = 0.0f;
#pragma unroll
  for (int v = 0; v < 8; ++v) { mrow[v] = -1e30f; lrow[v] = 0.0f; }

  const float scale = 0.08838834764831845f;  // 1/sqrt(128)
  const int nkb = (q_base + 16 + 31) / 32;   // 32-key blocks (causal bound)

  for (int kb = 0; kb < nkb; ++kb) {
    const int k0 = kb * 32;

    // ---- batch-issue all K and V fragment loads (independent registers) ---
    v16bf kf0[4], kf1[4], vf[8];
#pragma unroll
    for (int c = 0; c < 4; ++c) {
      kf0[c] = load_frag_b(Kbase, k0,      HD, c * 32, lane);
      kf1[c] = load_frag_b(Kbase, k0 + 16, HD, c * 32, lane);
    }
#pragma unroll
    for (int d = 0; d < 8; ++d)
      vf[d] = load_frag_b(Vbase, d * 16, SEQL, k0, lane);

    // ---- score tiles: S = Q K^T (two 16x16 tiles over 32 keys) -----------
    v8f s0, s1;
#pragma unroll
    for (int v = 0; v < 8; ++v) { s0[v] = 0.0f; s1[v] = 0.0f; }
#pragma unroll
    for (int c = 0; c < 4; ++c) {
      s0 = wmma_bf16(qf[c], kf0[c], s0);
      s1 = wmma_bf16(qf[c], kf1[c], s1);
    }

    // ---- scale + causal mask + online softmax (rows in 16-lane halves) ---
    const int kc0 = k0 + r, kc1 = k0 + 16 + r;
    float p0[8], p1[8], alpha[8];
#pragma unroll
    for (int v = 0; v < 8; ++v) {
      const int qi = q_base + half * 8 + v;
      float a = s0[v] * scale;
      float bb = s1[v] * scale;
      if (kc0 > qi) a = -1e30f;
      if (kc1 > qi) bb = -1e30f;
      float rm = fmaxf(a, bb);
      rm = fmaxf(rm, __shfl_xor(rm, 1, 32));
      rm = fmaxf(rm, __shfl_xor(rm, 2, 32));
      rm = fmaxf(rm, __shfl_xor(rm, 4, 32));
      rm = fmaxf(rm, __shfl_xor(rm, 8, 32));
      const float mn = fmaxf(mrow[v], rm);
      alpha[v] = __expf(mrow[v] - mn);
      mrow[v] = mn;
      const float e0 = __expf(a - mn);
      const float e1 = __expf(bb - mn);
      p0[v] = e0; p1[v] = e1;
      float rs = e0 + e1;
      rs += __shfl_xor(rs, 1, 32);
      rs += __shfl_xor(rs, 2, 32);
      rs += __shfl_xor(rs, 4, 32);
      rs += __shfl_xor(rs, 8, 32);
      lrow[v] = lrow[v] * alpha[v] + rs;
    }

#pragma unroll
    for (int d = 0; d < 8; ++d)
#pragma unroll
      for (int v = 0; v < 8; ++v) o[d][v] *= alpha[v];

    // ---- C-layout P -> row-major LDS tile (16 rows x 32 keys, bf16) ------
    __syncthreads();
#pragma unroll
    for (int v = 0; v < 8; ++v) {
      const int m = half * 8 + v;
      pl[m * 32 + r]      = (__bf16)p0[v];
      pl[m * 32 + 16 + r] = (__bf16)p1[v];
    }
    __syncthreads();

    const v16bf paf = load_frag_a(pl, 0, 32, 0, lane);
#pragma unroll
    for (int d = 0; d < 8; ++d)
      o[d] = wmma_bf16(paf, vf[d], o[d]);
    __syncthreads();
  }

  // normalize and write to [B,S,H,HD] bf16 for the output projection
#pragma unroll
  for (int v = 0; v < 8; ++v) {
    const int m = half * 8 + v;
    const float inv = 1.0f / lrow[v];
    const size_t row = (((size_t)b * SEQL + q_base + m) * NH + h) * HD;
#pragma unroll
    for (int d = 0; d < 8; ++d)
      Oout[row + d * 16 + r] = (__bf16)(o[d][v] * inv);
  }
}

// ---------------------------------------------------------------------------
extern "C" void kernel_launch(void* const* d_in, const int* in_sizes, int n_in,
                              void* d_out, int out_size, void* d_ws,
                              size_t ws_size, hipStream_t stream) {
  const float* x  = (const float*)d_in[0];
  const float* fc = (const float*)d_in[1];
  const float* fs = (const float*)d_in[2];
  const float* wq = (const float*)d_in[3];
  const float* wk = (const float*)d_in[4];
  const float* wv = (const float*)d_in[5];
  const float* wo = (const float*)d_in[6];
  float* out = (float*)d_out;

  char* ws = (char*)d_ws;
  const size_t MB = 1024u * 1024u;
  __bf16* xb   = (__bf16*)(ws + 0);          // 16 MiB  x  bf16 [4096,2048]
  __bf16* wqb  = (__bf16*)(ws + 16 * MB);    //  8 MiB
  __bf16* wkb  = (__bf16*)(ws + 24 * MB);    //  4 MiB
  __bf16* wvb  = (__bf16*)(ws + 28 * MB);    //  4 MiB
  __bf16* wob  = (__bf16*)(ws + 32 * MB);    //  8 MiB
  __bf16* qtmp = (__bf16*)(ws + 40 * MB);    // 16 MiB  [B,S,H,HD]
  __bf16* ktmp = (__bf16*)(ws + 56 * MB);    //  8 MiB  [B,S,KVH,HD]
  __bf16* vtmp = (__bf16*)(ws + 64 * MB);    //  8 MiB
  __bf16* Qh   = (__bf16*)(ws + 72 * MB);    // 16 MiB  [B,H,S,HD]
  __bf16* Kh   = (__bf16*)(ws + 88 * MB);    //  8 MiB  [B,KVH,S,HD]
  __bf16* Vt   = (__bf16*)(ws + 96 * MB);    //  8 MiB  [B,KVH,HD,S]
  __bf16* attn = (__bf16*)(ws + 40 * MB);    // reuse qtmp after RoPE

  const int M = BSZ * SEQL;  // 4096

  // 1) fp32 -> bf16 casts
  cast_f32_bf16<<<2048, 256, 0, stream>>>(x,  xb,  (size_t)M * DIM);
  cast_f32_bf16<<<1024, 256, 0, stream>>>(wq, wqb, (size_t)DIM * DIM);
  cast_f32_bf16<<<512,  256, 0, stream>>>(wk, wkb, (size_t)NKV * HD * DIM);
  cast_f32_bf16<<<512,  256, 0, stream>>>(wv, wvb, (size_t)NKV * HD * DIM);
  cast_f32_bf16<<<1024, 256, 0, stream>>>(wo, wob, (size_t)DIM * DIM);

  // 2) QKV projections (bf16 WMMA, NT)
  gemm_nt<false><<<dim3(DIM / 128, M / 128), 256, 0, stream>>>(
      xb, wqb, qtmp, M, DIM, DIM);
  gemm_nt<false><<<dim3(NKV * HD / 128, M / 128), 256, 0, stream>>>(
      xb, wkb, ktmp, M, NKV * HD, DIM);
  gemm_nt<false><<<dim3(NKV * HD / 128, M / 128), 256, 0, stream>>>(
      xb, wvb, vtmp, M, NKV * HD, DIM);

  // 3) RoPE + head-major relayout; V transpose for NT PV GEMM
  rope_relayout<<<4096, 256, 0, stream>>>(qtmp, fc, fs, Qh, NH);
  rope_relayout<<<2048, 256, 0, stream>>>(ktmp, fc, fs, Kh, NKV);
  transpose_v<<<4096, 256, 0, stream>>>(vtmp, Vt);

  // 4) causal flash attention (one wave per 16-query tile per head)
  flash_attn<<<BSZ * NH * (SEQL / 16), 32, 0, stream>>>(Qh, Kh, Vt, attn);

  // 5) output projection -> fp32 d_out
  gemm_nt<true><<<dim3(DIM / 128, M / 128), 256, 0, stream>>>(
      attn, wob, out, M, DIM, DIM);
}
